// GNNClassifier_74878459838708
// MI455X (gfx1250) — compile-verified, matrix operand order checked by reading, not run on hardware
//
#include <hip/hip_runtime.h>
#include <hip/hip_bf16.h>

typedef __bf16 bf16;
typedef __attribute__((ext_vector_type(16))) __bf16 v16bf;
typedef __attribute__((ext_vector_type(8)))  __bf16 v8bf;
typedef __attribute__((ext_vector_type(8)))  float   v8f;

#define NEG_SLOPE 0.2f
#define CH 64

static __device__ __forceinline__ unsigned f2ord(float f) {
    unsigned u = __float_as_uint(f);
    return (u & 0x80000000u) ? ~u : (u | 0x80000000u);
}
static __device__ __forceinline__ float ord2f(unsigned v) {
    return __uint_as_float((v & 0x80000000u) ? (v & 0x7fffffffu) : ~v);
}
static __device__ __forceinline__ float lrelu(float x) {
    return x > 0.0f ? x : NEG_SLOPE * x;
}

// ---------------- conversion kernels ----------------

__global__ void cvt_bf16_kernel(const float* __restrict__ s, bf16* __restrict__ d, long long n) {
    long long i = (long long)blockIdx.x * blockDim.x + threadIdx.x;
    if (i < n) d[i] = (bf16)s[i];
}

// W [K, Nn] row-major (f32) -> Wt [Nn, K] row-major (bf16)
__global__ void cvt_tr_bf16_kernel(const float* __restrict__ W, bf16* __restrict__ Wt,
                                   int K, int Nn) {
    int i = blockIdx.x * blockDim.x + threadIdx.x;
    if (i >= K * Nn) return;
    int k = i / Nn, n = i % Nn;
    Wt[(size_t)n * K + k] = (bf16)W[i];
}

__global__ void fill_u32_kernel(unsigned* __restrict__ p, unsigned v, long long n) {
    long long i = (long long)blockIdx.x * blockDim.x + threadIdx.x;
    if (i < n) p[i] = v;
}
__global__ void fill_f32_kernel(float* __restrict__ p, float v, long long n) {
    long long i = (long long)blockIdx.x * blockDim.x + threadIdx.x;
    if (i < n) p[i] = v;
}

// ---------------- WMMA GEMM: C[M,Nn] = A[M,K]bf16 * B, with B given transposed Bt[Nn,K] ----------------
// One wave32 computes one 16x16 tile of C with v_wmma_f32_16x16x32_bf16 over K.
// A-fragment per CDNA5 layout: lane&15 = row M, lane>>4 selects K-halves; each lane
// holds two contiguous 8-element bf16 runs (K = k0 + half*8 + [0..7] and +16).
// Bt rows are the columns of B, so the B-fragment uses the identical addressing.
__global__ __launch_bounds__(256)
void wmma_gemm_bf16_kernel(const bf16* __restrict__ A, const bf16* __restrict__ Bt,
                           float* __restrict__ C, int M, int Nn, int K) {
    const int lane  = threadIdx.x & 31;
    const int wave  = threadIdx.x >> 5;
    const int tilesM = (M + 15) >> 4;
    const int tilesN = Nn >> 4;
    const int wid   = blockIdx.x * 8 + wave;           // uniform per wave
    if (wid >= tilesM * tilesN) return;                // uniform exit, EXEC stays full for WMMA
    const int tm = wid / tilesN;
    const int tn = wid % tilesN;

    const int half = lane >> 4;
    const int l15  = lane & 15;
    int arow = tm * 16 + l15;
    if (arow >= M) arow = M - 1;                       // clamp (never taken for M%16==0)
    const bf16* ap = A  + (size_t)arow * K + half * 8;
    const bf16* bp = Bt + (size_t)(tn * 16 + l15) * K + half * 8;

    v8f acc = {};
    for (int k0 = 0; k0 < K; k0 += 32) {
        v8bf a_lo = *(const v8bf*)(ap + k0);
        v8bf a_hi = *(const v8bf*)(ap + k0 + 16);
        v8bf b_lo = *(const v8bf*)(bp + k0);
        v8bf b_hi = *(const v8bf*)(bp + k0 + 16);
        v16bf af = __builtin_shufflevector(a_lo, a_hi, 0,1,2,3,4,5,6,7,8,9,10,11,12,13,14,15);
        v16bf bf = __builtin_shufflevector(b_lo, b_hi, 0,1,2,3,4,5,6,7,8,9,10,11,12,13,14,15);
        acc = __builtin_amdgcn_wmma_f32_16x16x32_bf16(
            /*neg_a=*/false, af, /*neg_b=*/false, bf,
            /*c_mod=*/(short)0, acc, /*reuse_a=*/false, /*reuse_b=*/false);
    }

    // C/D layout: VGPR r holds row (r + 8*half), lane&15 is the column.
    const int col = tn * 16 + l15;
#pragma unroll
    for (int r = 0; r < 8; ++r) {
        int m = tm * 16 + half * 8 + r;
        if (m < M) C[(size_t)m * Nn + col] = acc[r];
    }
}

// ---------------- attention coefficients: a_src[n,h] = <h[n,h,:], att_src[h,:]> ----------------
__global__ void attn_coef_kernel(const float* __restrict__ h,
                                 const float* __restrict__ att_s,
                                 const float* __restrict__ att_d,
                                 float* __restrict__ as, float* __restrict__ ad,
                                 int N, int H) {
    int i = blockIdx.x * blockDim.x + threadIdx.x;
    if (i >= N * H) return;
    int n = i / H, hh = i % H;
    const float* hp = h + ((size_t)n * H + hh) * CH;
    const float* ws = att_s + hh * CH;
    const float* wd = att_d + hh * CH;
    float s = 0.f, d = 0.f;
#pragma unroll 8
    for (int c = 0; c < CH; ++c) { float v = hp[c]; s += v * ws[c]; d += v * wd[c]; }
    as[i] = s; ad[i] = d;
}

// ---------------- edge softmax passes (edges + implicit self loops) ----------------
__global__ void edge_max_kernel(const int* __restrict__ src, const int* __restrict__ dst,
                                int E, int N, int H,
                                const float* __restrict__ as, const float* __restrict__ ad,
                                unsigned* __restrict__ m) {
    int e = blockIdx.x * blockDim.x + threadIdx.x;
    if (e >= E + N) return;
    int s = (e < E) ? src[e] : (e - E);
    int d = (e < E) ? dst[e] : (e - E);
    for (int h = 0; h < H; ++h) {
        float v = lrelu(as[s * H + h] + ad[d * H + h]);
        atomicMax(&m[d * H + h], f2ord(v));
    }
}

__global__ void edge_sum_kernel(const int* __restrict__ src, const int* __restrict__ dst,
                                int E, int N, int H,
                                const float* __restrict__ as, const float* __restrict__ ad,
                                const unsigned* __restrict__ m, float* __restrict__ sm) {
    int e = blockIdx.x * blockDim.x + threadIdx.x;
    if (e >= E + N) return;
    int s = (e < E) ? src[e] : (e - E);
    int d = (e < E) ? dst[e] : (e - E);
    for (int h = 0; h < H; ++h) {
        float v = lrelu(as[s * H + h] + ad[d * H + h]);
        atomicAdd(&sm[d * H + h], expf(v - ord2f(m[d * H + h])));
    }
}

// 64 threads per edge (one per channel), loop heads: out[dst,h,c] += h[src,h,c] * alpha
__global__ __launch_bounds__(256)
void edge_agg_kernel(const int* __restrict__ src, const int* __restrict__ dst,
                     int E, int N, int H,
                     const float* __restrict__ as, const float* __restrict__ ad,
                     const unsigned* __restrict__ m, const float* __restrict__ sm,
                     const float* __restrict__ hfeat, float* __restrict__ outp) {
    long long gid = (long long)blockIdx.x * blockDim.x + threadIdx.x;
    long long e = gid >> 6;
    int c = (int)(gid & 63);
    if (e >= (long long)(E + N)) return;
    int s = (e < (long long)E) ? src[e] : (int)(e - E);
    int d = (e < (long long)E) ? dst[e] : (int)(e - E);
    const float* hp = hfeat + (size_t)s * H * CH;
    __builtin_prefetch(hp, 0, 0);
    for (int h = 0; h < H; ++h) {
        float v = lrelu(as[s * H + h] + ad[d * H + h]);
        float alpha = expf(v - ord2f(m[d * H + h])) / sm[d * H + h];
        atomicAdd(&outp[((size_t)d * H + h) * CH + c], hp[h * CH + c] * alpha);
    }
}

// mean over heads + bias + relu; optional bf16 copy for the next GEMM
__global__ void finalize_kernel(const float* __restrict__ agg, const float* __restrict__ bias,
                                float* __restrict__ hf, bf16* __restrict__ hb, int N, int H) {
    int i = blockIdx.x * blockDim.x + threadIdx.x;
    if (i >= N * CH) return;
    int n = i >> 6, c = i & 63;
    float acc = 0.f;
    for (int h = 0; h < H; ++h) acc += agg[((size_t)n * H + h) * CH + c];
    float v = acc * (1.0f / (float)H) + bias[c];
    v = v > 0.f ? v : 0.f;
    hf[i] = v;
    if (hb) hb[i] = (bf16)v;
}

__global__ void pool_max_kernel(const float* __restrict__ h2, const int* __restrict__ batch,
                                unsigned* __restrict__ pooled, int N) {
    int i = blockIdx.x * blockDim.x + threadIdx.x;
    if (i >= N * CH) return;
    int n = i >> 6, c = i & 63;
    atomicMax(&pooled[batch[n] * CH + c], f2ord(h2[i]));
}

__global__ void final_linear_kernel(const unsigned* __restrict__ pooled,
                                    const float* __restrict__ w, const float* __restrict__ b,
                                    float* __restrict__ out, int G) {
    int g = blockIdx.x * blockDim.x + threadIdx.x;
    if (g >= G) return;
    float acc = 0.f;
#pragma unroll 8
    for (int c = 0; c < CH; ++c) acc += ord2f(pooled[g * CH + c]) * w[c];
    out[g] = acc + b[0];
}

// ---------------- host orchestration ----------------

static inline int cdiv_i(long long a, long long b) { return (int)((a + b - 1) / b); }

extern "C" void kernel_launch(void* const* d_in, const int* in_sizes, int n_in,
                              void* d_out, int out_size, void* d_ws, size_t ws_size,
                              hipStream_t stream) {
    const float* x        = (const float*)d_in[0];
    const int*   eidx     = (const int*)  d_in[1];
    const int*   batch    = (const int*)  d_in[2];
    const float* W1       = (const float*)d_in[3];
    const float* att_s1   = (const float*)d_in[4];
    const float* att_d1   = (const float*)d_in[5];
    const float* b1       = (const float*)d_in[6];
    const float* W2       = (const float*)d_in[7];
    const float* att_s2   = (const float*)d_in[8];
    const float* att_d2   = (const float*)d_in[9];
    const float* b2       = (const float*)d_in[10];
    const float* lin_w    = (const float*)d_in[11];
    const float* lin_b    = (const float*)d_in[12];

    const int Fin = 768;
    const int N   = in_sizes[0] / Fin;
    const int E   = in_sizes[1] / 2;
    const int H1  = in_sizes[4] / CH;      // 4
    const int H2  = in_sizes[8] / CH;      // 6
    const int C1  = H1 * CH;               // 256
    const int C2  = H2 * CH;               // 384
    const int G   = 256;
    const int Hmax = H1 > H2 ? H1 : H2;
    const unsigned ENC_NEG_INF = 0x007fffffu;   // f2ord(-inf)

    const int* e_src = eidx;
    const int* e_dst = eidx + E;

    // ---- workspace bump allocator (with aliasing) ----
    char* ws = (char*)d_ws;
    size_t off = 0;
    auto alloc = [&](size_t bytes) -> char* {
        char* p = ws + off;
        off += (bytes + 255) & ~(size_t)255;
        return p;
    };
    bf16*  x_bf  = (bf16*)alloc((size_t)N * Fin * sizeof(bf16));   // later aliased as h2raw
    bf16*  W1t   = (bf16*)alloc((size_t)C1 * Fin * sizeof(bf16));
    char*  big   = alloc(2 * (size_t)N * C1 * sizeof(float));      // h1raw+agg1, later agg2
    float* h1raw = (float*)big;
    float* agg1  = (float*)(big + (size_t)N * C1 * sizeof(float));
    float* as_b  = (float*)alloc((size_t)N * Hmax * sizeof(float));
    float* ad_b  = (float*)alloc((size_t)N * Hmax * sizeof(float));
    unsigned* m_b = (unsigned*)alloc((size_t)N * Hmax * sizeof(unsigned));
    float* s_b   = (float*)alloc((size_t)N * Hmax * sizeof(float));
    bf16*  h1bf  = (bf16*)alloc((size_t)N * CH * sizeof(bf16));
    float* h1f   = (float*)alloc((size_t)N * CH * sizeof(float));
    bf16*  W2t   = (bf16*)alloc((size_t)C2 * CH * sizeof(bf16));
    float* h2f   = (float*)alloc((size_t)N * CH * sizeof(float));
    unsigned* pooled = (unsigned*)alloc((size_t)G * CH * sizeof(unsigned));
    float* h2raw = (float*)x_bf;   // alias: x_bf dead after GEMM1, sizes match
    float* agg2  = (float*)big;    // alias: h1raw/agg1 dead after finalize-1

    const int TPB = 256;

    // ---- layer 1: feature transform via WMMA ----
    cvt_bf16_kernel<<<cdiv_i((long long)N * Fin, TPB), TPB, 0, stream>>>(x, x_bf, (long long)N * Fin);
    cvt_tr_bf16_kernel<<<cdiv_i((long long)Fin * C1, TPB), TPB, 0, stream>>>(W1, W1t, Fin, C1);
    {
        int tiles = ((N + 15) / 16) * (C1 / 16);
        wmma_gemm_bf16_kernel<<<cdiv_i(tiles, 8), TPB, 0, stream>>>(x_bf, W1t, h1raw, N, C1, Fin);
    }
    attn_coef_kernel<<<cdiv_i((long long)N * H1, TPB), TPB, 0, stream>>>(h1raw, att_s1, att_d1, as_b, ad_b, N, H1);
    fill_u32_kernel<<<cdiv_i((long long)N * H1, TPB), TPB, 0, stream>>>(m_b, ENC_NEG_INF, (long long)N * H1);
    fill_f32_kernel<<<cdiv_i((long long)N * H1, TPB), TPB, 0, stream>>>(s_b, 0.f, (long long)N * H1);
    fill_f32_kernel<<<cdiv_i((long long)N * C1, TPB), TPB, 0, stream>>>(agg1, 0.f, (long long)N * C1);
    edge_max_kernel<<<cdiv_i(E + N, TPB), TPB, 0, stream>>>(e_src, e_dst, E, N, H1, as_b, ad_b, m_b);
    edge_sum_kernel<<<cdiv_i(E + N, TPB), TPB, 0, stream>>>(e_src, e_dst, E, N, H1, as_b, ad_b, m_b, s_b);
    edge_agg_kernel<<<cdiv_i((long long)(E + N) * CH, TPB), TPB, 0, stream>>>(
        e_src, e_dst, E, N, H1, as_b, ad_b, m_b, s_b, h1raw, agg1);
    finalize_kernel<<<cdiv_i((long long)N * CH, TPB), TPB, 0, stream>>>(agg1, b1, h1f, h1bf, N, H1);

    // ---- layer 2 ----
    cvt_tr_bf16_kernel<<<cdiv_i((long long)CH * C2, TPB), TPB, 0, stream>>>(W2, W2t, CH, C2);
    {
        int tiles = ((N + 15) / 16) * (C2 / 16);
        wmma_gemm_bf16_kernel<<<cdiv_i(tiles, 8), TPB, 0, stream>>>(h1bf, W2t, h2raw, N, C2, CH);
    }
    attn_coef_kernel<<<cdiv_i((long long)N * H2, TPB), TPB, 0, stream>>>(h2raw, att_s2, att_d2, as_b, ad_b, N, H2);
    fill_u32_kernel<<<cdiv_i((long long)N * H2, TPB), TPB, 0, stream>>>(m_b, ENC_NEG_INF, (long long)N * H2);
    fill_f32_kernel<<<cdiv_i((long long)N * H2, TPB), TPB, 0, stream>>>(s_b, 0.f, (long long)N * H2);
    fill_f32_kernel<<<cdiv_i((long long)N * C2, TPB), TPB, 0, stream>>>(agg2, 0.f, (long long)N * C2);
    edge_max_kernel<<<cdiv_i(E + N, TPB), TPB, 0, stream>>>(e_src, e_dst, E, N, H2, as_b, ad_b, m_b);
    edge_sum_kernel<<<cdiv_i(E + N, TPB), TPB, 0, stream>>>(e_src, e_dst, E, N, H2, as_b, ad_b, m_b, s_b);
    edge_agg_kernel<<<cdiv_i((long long)(E + N) * CH, TPB), TPB, 0, stream>>>(
        e_src, e_dst, E, N, H2, as_b, ad_b, m_b, s_b, h2raw, agg2);
    finalize_kernel<<<cdiv_i((long long)N * CH, TPB), TPB, 0, stream>>>(agg2, b2, h2f, (bf16*)nullptr, N, H2);

    // ---- pooling + head ----
    fill_u32_kernel<<<cdiv_i((long long)G * CH, TPB), TPB, 0, stream>>>(pooled, ENC_NEG_INF, (long long)G * CH);
    pool_max_kernel<<<cdiv_i((long long)N * CH, TPB), TPB, 0, stream>>>(h2f, batch, pooled, N);
    final_linear_kernel<<<1, TPB, 0, stream>>>(pooled, lin_w, lin_b, (float*)d_out, G);

    (void)n_in; (void)out_size; (void)ws_size;
}
